// PointNetFeaturePropagation_36026185679270
// MI455X (gfx1250) — matrix operand order repaired and checked by reading, not compiled, not run on hardware
//
#include <hip/hip_runtime.h>
#include <stdint.h>

// Problem dims (fixed by reference setup_inputs)
#define B_    8
#define N_    8192
#define S_    2048
#define D1_   128
#define D2_   256
#define C0_   384      // D1 + D2 (GEMM1 K)
#define C1_   256      // layer0 out
#define C2_   128      // layer1 out
#define ROWS_ 65536    // B*N

typedef __bf16 bf16x16 __attribute__((ext_vector_type(16)));
typedef float  f32x8   __attribute__((ext_vector_type(8)));

union V16 { bf16x16 v; uint4 q[2]; };

// ---------------------------------------------------------------- utilities
__global__ void zero_f32(float* p, int n) {
  int i = blockIdx.x * blockDim.x + threadIdx.x;
  if (i < n) p[i] = 0.f;
}

__global__ void cvt_f32_bf16(const float* __restrict__ src,
                             __bf16* __restrict__ dst, int n) {
  int i = blockIdx.x * blockDim.x + threadIdx.x;
  if (i < n) dst[i] = (__bf16)src[i];
}

// ---------------------------------------------------- 3-NN interp + concat
// Grid: B_*(N_/256) blocks of 256 threads. Phase 1: one thread per query
// point scans all S=2048 reference points held in LDS (24 KB), keeping the
// 3 smallest squared distances. Phase 2: waves cooperatively gather the 3
// feature rows (coalesced 128B reads) and emit bf16 X = [points1 | interp].
__global__ void __launch_bounds__(256)
interp_build_x(const float* __restrict__ xyz1, const float* __restrict__ xyz2,
               const float* __restrict__ points1, const float* __restrict__ points2,
               __bf16* __restrict__ X)
{
  __shared__ float sxyz[S_ * 3];     // 24 KB
  __shared__ int   sidx[256 * 3];
  __shared__ float swgt[256 * 3];

  const int tid  = threadIdx.x;
  const int b    = blockIdx.x >> 5;        // 32 blocks per batch
  const int nblk = blockIdx.x & 31;

  const float* x2 = xyz2 + (size_t)b * S_ * 3;
  for (int i = tid; i < S_ * 3; i += 256) sxyz[i] = x2[i];
  __syncthreads();

  // ---- phase 1: brute-force top-3 nearest
  const size_t gn = (size_t)b * N_ + nblk * 256 + tid;
  const float px = xyz1[gn * 3 + 0];
  const float py = xyz1[gn * 3 + 1];
  const float pz = xyz1[gn * 3 + 2];
  float d0 = 3.4e38f, d1 = 3.4e38f, d2 = 3.4e38f;
  int   i0 = 0, i1 = 0, i2 = 0;
  for (int s = 0; s < S_; ++s) {
    float dx = px - sxyz[3 * s + 0];
    float dy = py - sxyz[3 * s + 1];
    float dz = pz - sxyz[3 * s + 2];
    float d  = dx * dx + dy * dy + dz * dz;
    if (d < d2) {
      if (d < d1) {
        if (d < d0) { d2 = d1; i2 = i1; d1 = d0; i1 = i0; d0 = d; i0 = s; }
        else        { d2 = d1; i2 = i1; d1 = d;  i1 = s; }
      } else        { d2 = d;  i2 = s; }
    }
  }
  float w0 = 1.f / (d0 + 1e-8f);
  float w1 = 1.f / (d1 + 1e-8f);
  float w2 = 1.f / (d2 + 1e-8f);
  float inv = 1.f / (w0 + w1 + w2);
  swgt[3 * tid + 0] = w0 * inv; swgt[3 * tid + 1] = w1 * inv; swgt[3 * tid + 2] = w2 * inv;
  sidx[3 * tid + 0] = i0;       sidx[3 * tid + 1] = i1;       sidx[3 * tid + 2] = i2;
  __syncthreads();

  // ---- phase 2: wave-cooperative coalesced gather + concat, bf16 out
  const int wid  = tid >> 5;
  const int lane = tid & 31;
  const float* P2 = points2 + (size_t)b * S_ * D2_;
  for (int p = wid; p < 256; p += 8) {
    const size_t row = (size_t)b * N_ + nblk * 256 + p;
    __bf16* xr = X + row * C0_;
    const float* p1 = points1 + row * D1_;
    #pragma unroll
    for (int c0 = 0; c0 < D1_; c0 += 32)
      xr[c0 + lane] = (__bf16)p1[c0 + lane];
    const float a0 = swgt[3 * p + 0], a1 = swgt[3 * p + 1], a2 = swgt[3 * p + 2];
    const float* r0 = P2 + (size_t)sidx[3 * p + 0] * D2_;
    const float* r1 = P2 + (size_t)sidx[3 * p + 1] * D2_;
    const float* r2 = P2 + (size_t)sidx[3 * p + 2] * D2_;
    #pragma unroll
    for (int c0 = 0; c0 < D2_; c0 += 32) {
      int c = c0 + lane;
      float v = a0 * r0[c] + a1 * r1[c] + a2 * r2[c];
      xr[D1_ + c] = (__bf16)v;
    }
  }
}

// --------------------------------------------------------- WMMA bf16 GEMM
// Y[r, o] = sum_k A[r, k] * W[o, k]     (W row-major [Cout, K] == B^T layout)
// One wave per 16x64 output tile: 4 accumulators, A fragment reused 4x.
// VGPR layouts per CDNA5 ISA 7.12.2:
//   A 16x32 bf16 : lane (m = lane&15, h = lane>>4); vgpr v holds K pair
//                  k = (v>>2)*16 + h*8 + (v&3)*2  -> two contiguous 16B loads
//                  at k-offsets {h*8, 16+h*8}.
//   B 32x16 bf16 : lane n = lane&15; same K order as A
//                  -> two contiguous 16B loads at k-offsets {h*8, 16+h*8}.
//   D 16x16 f32  : vgpr r -> row m = r + 8*h, col n = lane&15.
__global__ void __launch_bounds__(256)
gemm_bf16_wmma(const __bf16* __restrict__ A, const __bf16* __restrict__ W,
               float* __restrict__ Y, int K, int Cout, int ngrp)
{
  const int lane = threadIdx.x & 31;
  const int m = lane & 15;
  const int h = lane >> 4;
  const int wid   = blockIdx.x * 8 + (threadIdx.x >> 5);
  const int mt    = wid / ngrp;
  const int nbase = (wid % ngrp) * 64;

  const size_t arow = ((size_t)mt * 16 + m) * K;
  f32x8 acc[4] = {};

  for (int kk = 0; kk < K; kk += 32) {
    V16 a;
    a.q[0] = *(const uint4*)(A + arow + kk + h * 8);
    a.q[1] = *(const uint4*)(A + arow + kk + 16 + h * 8);
    // WGP-scope prefetch of the A tile two k-steps ahead (L2 -> WGP$): locality 3
    __builtin_prefetch(A + arow + kk + 64, 0, 3);   // global_prefetch_b8
    #pragma unroll
    for (int t = 0; t < 4; ++t) {
      const size_t wrow = (size_t)(nbase + t * 16 + m) * K;
      V16 bm;
      bm.q[0] = *(const uint4*)(W + wrow + kk + h * 8);
      bm.q[1] = *(const uint4*)(W + wrow + kk + 16 + h * 8);
      acc[t] = __builtin_amdgcn_wmma_f32_16x16x32_bf16(
          /*neg_a=*/false, a.v, /*neg_b=*/false, bm.v,
          /*c_mod=*/(short)0, acc[t], /*reuse_a=*/false, /*reuse_b=*/false);
    }
  }

  #pragma unroll
  for (int t = 0; t < 4; ++t) {
    const int col = nbase + t * 16 + m;
    #pragma unroll
    for (int r = 0; r < 8; ++r)
      Y[((size_t)mt * 16 + r + 8 * h) * Cout + col] = acc[t][r];
  }
}

// --------------------------------------------------------------- BN stats
// Grid: ROWS_/256 blocks of Cout threads; thread t owns channel t; reads
// are fully coalesced (consecutive lanes -> consecutive channels).
__global__ void bn_stats(const float* __restrict__ Y, float* __restrict__ st,
                         int Cout)
{
  const int o = threadIdx.x;
  const size_t r0 = (size_t)blockIdx.x * 256;
  float s = 0.f, s2 = 0.f;
  for (int r = 0; r < 256; ++r) {
    float v = Y[(r0 + r) * Cout + o];
    s += v; s2 += v * v;
  }
  atomicAdd(&st[o], s);
  atomicAdd(&st[Cout + o], s2);
}

// ------------------------------------------- BN finalize: per-channel a,b
// One thread per channel folds mean/var/gamma/beta into scale+shift so the
// big elementwise pass is a pure fma+relu stream. (The conv bias cancels
// exactly under BN, so it is omitted.)
__global__ void bn_finalize(float* __restrict__ st,
                            const float* __restrict__ gamma,
                            const float* __restrict__ beta, int Cout)
{
  const int o = threadIdx.x;
  const float invn = 1.f / (float)ROWS_;
  const float mean = st[o] * invn;
  const float var  = st[Cout + o] * invn - mean * mean;
  const float sc = gamma[o] * rsqrtf(var + 1e-5f);
  st[2 * Cout + o] = sc;
  st[3 * Cout + o] = beta[o] - mean * sc;
}

// ------------------------------------------------- BN normalize + ReLU
// Pure streaming pass: y*scale + shift, relu, store. Template picks bf16
// (next-layer input) or f32 (d_out).
template <typename OutT>
__global__ void bn_relu(const float* __restrict__ Y, const float* __restrict__ st,
                        OutT* __restrict__ out, int Cout)
{
  const size_t i = (size_t)blockIdx.x * blockDim.x + threadIdx.x;
  const int o = (int)(i & (size_t)(Cout - 1));   // Cout is a power of two
  const float sc = st[2 * Cout + o];
  const float sh = st[3 * Cout + o];
  float v = Y[i] * sc + sh;
  out[i] = (OutT)(v > 0.f ? v : 0.f);
}

// ------------------------------------------------------------------ launch
extern "C" void kernel_launch(void* const* d_in, const int* in_sizes, int n_in,
                              void* d_out, int out_size, void* d_ws, size_t ws_size,
                              hipStream_t stream) {
  const float* xyz1    = (const float*)d_in[0];
  const float* xyz2    = (const float*)d_in[1];
  const float* points1 = (const float*)d_in[2];
  const float* points2 = (const float*)d_in[3];
  const float* w0   = (const float*)d_in[4];
  const float* g0   = (const float*)d_in[6];
  const float* be0  = (const float*)d_in[7];
  const float* w1   = (const float*)d_in[8];
  const float* g1   = (const float*)d_in[10];
  const float* be1  = (const float*)d_in[11];

  // Workspace layout (regions aliased across pipeline stages):
  //   [W0b bf16 256x384][W1b bf16 128x256][stats 8 KB][X region][Y region]
  // stats: S0 = {sum[256], sumsq[256], sc[256], sh[256]},
  //        S1 = {sum[128], sumsq[128], sc[128], sh[128]}
  char* ws = (char*)d_ws;
  __bf16* W0b = (__bf16*)ws;                                   // 196608 B
  __bf16* W1b = (__bf16*)(ws + 196608);                        //  65536 B
  float*  S0  = (float*)(ws + 196608 + 65536);                 //   8192 B
  float*  S1  = S0 + 1024;
  __bf16* X   = (__bf16*)(ws + 196608 + 65536 + 8192);         // 50.3 MB (X1, then X2)
  float*  Yb  = (float*)(ws + 196608 + 65536 + 8192
                         + (size_t)ROWS_ * C0_ * 2);           // 67.1 MB (Y1, then Y2)

  // 1. weights -> bf16, zero stat accumulators
  cvt_f32_bf16<<<(C1_ * C0_ + 255) / 256, 256, 0, stream>>>(w0, W0b, C1_ * C0_);
  cvt_f32_bf16<<<(C2_ * C1_ + 255) / 256, 256, 0, stream>>>(w1, W1b, C2_ * C1_);
  zero_f32<<<6, 256, 0, stream>>>(S0, 1536);   // covers S0 sums + S1 sums

  // 2. interpolation + concat -> bf16 X1 [ROWS_, 384]
  interp_build_x<<<B_ * (N_ / 256), 256, 0, stream>>>(xyz1, xyz2, points1, points2, X);

  // 3. layer 0: GEMM (384 -> 256), BN stats, normalize+ReLU -> bf16 X2
  gemm_bf16_wmma<<<(ROWS_ / 16) * (C1_ / 64) / 8, 256, 0, stream>>>(X, W0b, Yb, C0_, C1_, C1_ / 64);
  bn_stats<<<ROWS_ / 256, C1_, 0, stream>>>(Yb, S0, C1_);
  bn_finalize<<<1, C1_, 0, stream>>>(S0, g0, be0, C1_);
  bn_relu<__bf16><<<(size_t)ROWS_ * C1_ / 256, 256, 0, stream>>>(Yb, S0, X, C1_);

  // 4. layer 1: GEMM (256 -> 128), BN stats, normalize+ReLU -> f32 d_out
  gemm_bf16_wmma<<<(ROWS_ / 16) * (C2_ / 64) / 8, 256, 0, stream>>>(X, W1b, Yb, C1_, C2_, C2_ / 64);
  bn_stats<<<ROWS_ / 256, C2_, 0, stream>>>(Yb, S1, C2_);
  bn_finalize<<<1, C2_, 0, stream>>>(S1, g1, be1, C2_);
  bn_relu<float><<<(size_t)ROWS_ * C2_ / 256, 256, 0, stream>>>(Yb, S1, (float*)d_out, C2_);
}
